// ScaledDotProductAttention_3925600108937
// MI455X (gfx1250) — matrix operand order repaired and checked by reading, not compiled, not run on hardware
//
#include <hip/hip_runtime.h>

// ScaledDotProductAttention with post-softmax masking + renormalization.
// B=16, L=2048, D=64, fp32 in/out. Outputs: output [B,L,D] then attn [B,L,L].
//
// Memory-bound (~0.56 GB HBM: 268MB attn write + 268MB mask read + QKV):
//  - one WG (256 thr = 8 waves) per (batch, 16-query-row) tile
//  - 16x2048 fp32 score tile in LDS; attn written to HBM exactly once
//  - QK^T and P*V on v_wmma_f32_16x16x32_bf16; K/V stay L2-resident (1MB/batch)
//  - float4/int4 sweeps for the mask/renorm streams; global_prefetch_b8 to
//    overlap the HBM-cold mask stream with the LDS softmax passes

#define LQ 2048
#define DDIM 64
#define SSTRIDE 2052            // 2048 + 4 pad -> conflict-free column gathers
#define EPSILON 1e-8f
#define INVTEMP 0.125f          // 1 / sqrt(64)

typedef __attribute__((ext_vector_type(16))) __bf16 v16bf;
typedef __attribute__((ext_vector_type(8)))  float  v8f;

__global__ __launch_bounds__(256)
void attn_postmask_kernel(const float* __restrict__ Q,
                          const float* __restrict__ K,
                          const float* __restrict__ V,
                          const int*   __restrict__ Mask,   // [B,L,L], nonzero => zero out
                          float* __restrict__ Out,          // [B,L,D]
                          float* __restrict__ AttnOut)      // [B,L,L]
{
    extern __shared__ float smem[];
    float* sS    = smem;                  // [16][SSTRIDE] score / prob tile
    float* sRed  = sS + 16 * SSTRIDE;     // [16][16] partial reductions
    float* sRow  = sRed + 16 * 16;        // [16]     per-row reduced value
    float* sPart = sRow + 16;             // [4][16][16] split-K partials for P*V

    const int qtile = blockIdx.x;         // 0..127
    const int b     = blockIdx.y;         // 0..15
    const int t     = threadIdx.x;        // 0..255
    const int wave  = t >> 5;             // 0..7
    const int lane  = t & 31;
    const int lhi   = lane >> 4;          // half-wave select
    const int llo   = lane & 15;

    const int q0 = qtile * 16;
    const float* Qb = Q + ((size_t)b * LQ + q0) * DDIM;
    const float* Kb = K + (size_t)b * LQ * DDIM;
    const float* Vb = V + (size_t)b * LQ * DDIM;

    // ---- Build the two Q A-fragments once (reused for all key tiles).
    // A(16x32): lane(m=llo, lhi); slot s -> k = 16*lhi + s
    v16bf aq0, aq1;
#pragma unroll
    for (int s = 0; s < 16; ++s) {
        const int k = lhi * 16 + s;
        aq0[s] = (__bf16)(Qb[llo * DDIM + k]      * INVTEMP);
        aq1[s] = (__bf16)(Qb[llo * DDIM + 32 + k] * INVTEMP);
    }

    // ---- Phase 1: S = (Q K^T)/temp. Each of 8 waves takes every 8th key tile.
    for (int tile = wave; tile < LQ / 16; tile += 8) {
        const int key0 = tile * 16;
        // B(32x16): lane(n=llo, lhi); slot s -> k = 16*lhi + s
        // element(k,n) = K[key0+n][d0+k] -> 16 consecutive floats per lane.
        const float* kp = Kb + (size_t)(key0 + llo) * DDIM + lhi * 16;
        if (tile + 8 < LQ / 16) {         // prefetch this wave's next K tile
            const float* kpn = kp + 128 * DDIM;
            __builtin_prefetch(kpn, 0, 1);
            __builtin_prefetch(kpn + 32, 0, 1);
        }
        v16bf b0, b1;
#pragma unroll
        for (int s = 0; s < 16; ++s) {
            b0[s] = (__bf16)kp[s];        // d0 = 0
            b1[s] = (__bf16)kp[32 + s];   // d0 = 32
        }
        v8f c = {};
        c = __builtin_amdgcn_wmma_f32_16x16x32_bf16(false, aq0, false, b0,
                                                    (short)0, c, false, false);
        c = __builtin_amdgcn_wmma_f32_16x16x32_bf16(false, aq1, false, b1,
                                                    (short)0, c, false, false);
        // C/D: vgpr r -> row m = r + 8*lhi, col n = llo
#pragma unroll
        for (int r = 0; r < 8; ++r)
            sS[(r + 8 * lhi) * SSTRIDE + key0 + llo] = c[r];
    }

    // ---- Softmax / mask / renorm: 16 threads per row, 128-element chunks.
    const int r   = t >> 4;               // row 0..15
    const int sub = t & 15;
    float*        rowp = sS + r * SSTRIDE + sub * 128;
    const int*    mrow = Mask + (((size_t)b * LQ) + (q0 + r)) * LQ + sub * 128;

    // Kick the HBM-cold mask stream now; it lands during the LDS-only passes.
    {
        const char* mp = (const char*)mrow;
        __builtin_prefetch(mp, 0, 0);
        __builtin_prefetch(mp + 128, 0, 0);
        __builtin_prefetch(mp + 256, 0, 0);
        __builtin_prefetch(mp + 384, 0, 0);
    }
    __syncthreads();

    // 2a: row max (inputs finite -> plain compare, no NaN bookkeeping)
    float4* rp4 = (float4*)rowp;
    float mx = -3.4e38f;
    for (int i = 0; i < 32; ++i) {
        const float4 x = rp4[i];
        const float m01 = (x.x > x.y) ? x.x : x.y;
        const float m23 = (x.z > x.w) ? x.z : x.w;
        const float m   = (m01 > m23) ? m01 : m23;
        mx = (m > mx) ? m : mx;
    }
    sRed[r * 16 + sub] = mx;
    __syncthreads();
    if (sub == 0) {
        float m2 = sRed[r * 16];
        for (int j = 1; j < 16; ++j) {
            const float v2 = sRed[r * 16 + j];
            m2 = (v2 > m2) ? v2 : m2;
        }
        sRow[r] = m2;
    }
    __syncthreads();
    mx = sRow[r];

    // 2b: exp(S - max) in place + row sum
    float acc = 0.f;
    for (int i = 0; i < 32; ++i) {
        float4 x = rp4[i];
        x.x = __expf(x.x - mx); x.y = __expf(x.y - mx);
        x.z = __expf(x.z - mx); x.w = __expf(x.w - mx);
        acc += (x.x + x.y) + (x.z + x.w);
        rp4[i] = x;
    }
    sRed[r * 16 + sub] = acc;
    __syncthreads();
    if (sub == 0) {
        float s2 = 0.f;
        for (int j = 0; j < 16; ++j) s2 += sRed[r * 16 + j];
        sRow[r] = s2;
    }
    __syncthreads();
    const float invL = 1.0f / sRow[r];

    // 3: post-softmax mask + masked row sum (p = e*invL; a = mask ? 0 : p)
    const int4* m4 = (const int4*)mrow;
    float macc = 0.f;
    for (int i = 0; i < 32; ++i) {
        const int4 mm = m4[i];
        float4 x = rp4[i];
        x.x = mm.x ? 0.0f : x.x * invL;
        x.y = mm.y ? 0.0f : x.y * invL;
        x.z = mm.z ? 0.0f : x.z * invL;
        x.w = mm.w ? 0.0f : x.w * invL;
        macc += (x.x + x.y) + (x.z + x.w);
        rp4[i] = x;
    }
    sRed[r * 16 + sub] = macc;
    __syncthreads();
    if (sub == 0) {
        float s2 = 0.f;
        for (int j = 0; j < 16; ++j) s2 += sRed[r * 16 + j];
        sRow[r] = s2;
    }
    __syncthreads();
    const float factor = 1.0f / (sRow[r] + EPSILON);

    // 4: renormalize, keep in LDS for P*V, single vectorized HBM write of attn
    float4* a4 = (float4*)(AttnOut + (((size_t)b * LQ) + (q0 + r)) * LQ + sub * 128);
    for (int i = 0; i < 32; ++i) {
        float4 x = rp4[i];
        x.x *= factor; x.y *= factor; x.z *= factor; x.w *= factor;
        rp4[i] = x;
        a4[i]  = x;
    }
    __syncthreads();

    // ---- Phase 6: O = P * V, split-K across wave pairs.
    // wave&3 -> output column block, wave>>2 -> half of the 2048 contraction.
    const int cb = (wave & 3) * 16;
    const int kh = wave >> 2;
    const int kbeg = kh * (LQ / 2);
    const int kend = kbeg + (LQ / 2);
    v8f oacc = {};
    for (int k0 = kbeg; k0 < kend; k0 += 32) {
        if (k0 + 32 < kend)               // prefetch next V chunk (L2-resident)
            __builtin_prefetch(Vb + (size_t)(k0 + 32 + lhi * 16) * DDIM + cb + llo, 0, 1);
        v16bf ap, bv;
#pragma unroll
        for (int s = 0; s < 16; ++s) {
            const int kk = lhi * 16 + s;
            ap[s] = (__bf16)sS[llo * SSTRIDE + k0 + kk];                       // P[m][k0+kk]
            bv[s] = (__bf16)Vb[(size_t)(k0 + lhi * 16 + s) * DDIM + cb + llo]; // V[k][cb+n]
        }
        oacc = __builtin_amdgcn_wmma_f32_16x16x32_bf16(false, ap, false, bv,
                                                       (short)0, oacc, false, false);
    }
    // Upper-half waves park partials in LDS; lower-half waves combine + store.
    if (kh == 1) {
#pragma unroll
        for (int rr = 0; rr < 8; ++rr)
            sPart[(wave - 4) * 256 + (rr + 8 * lhi) * 16 + llo] = oacc[rr];
    }
    __syncthreads();
    if (kh == 0) {
        float* outp = Out + ((size_t)b * LQ + q0) * DDIM + cb;
#pragma unroll
        for (int rr = 0; rr < 8; ++rr)
            outp[(rr + 8 * lhi) * DDIM + llo] =
                oacc[rr] + sPart[wave * 256 + (rr + 8 * lhi) * 16 + llo];
    }
}

extern "C" void kernel_launch(void* const* d_in, const int* in_sizes, int n_in,
                              void* d_out, int out_size, void* d_ws, size_t ws_size,
                              hipStream_t stream) {
    const float* q    = (const float*)d_in[0];
    const float* k    = (const float*)d_in[1];
    const float* v    = (const float*)d_in[2];
    const int*   mask = (const int*)d_in[3];   // bool mask surfaced as int32

    float* out  = (float*)d_out;                       // [16,2048,64]
    float* attn = out + (size_t)16 * LQ * DDIM;        // [16,2048,2048]

    const dim3 grid(LQ / 16, 16);                      // 128 q-tiles x 16 batches
    const dim3 block(256);                             // 8 waves (wave32)
    const size_t smem =
        (size_t)(16 * SSTRIDE + 16 * 16 + 16 + 4 * 16 * 16) * sizeof(float);

    attn_postmask_kernel<<<grid, block, smem, stream>>>(q, k, v, mask, out, attn);
}
